// Dinov3ViTAttention_62766652064228
// MI455X (gfx1250) — matrix operand order repaired and checked by reading, not compiled, not run on hardware
//
#include <hip/hip_runtime.h>
#include <math.h>

#define B_    16
#define T_    1029
#define P_    1024
#define NPREF 5
#define H_    1024
#define NH_   16
#define HD_   64
#define MT_   (B_ * T_)   // 16464 rows

typedef __attribute__((ext_vector_type(16))) __bf16 v16bf;
typedef __attribute__((ext_vector_type(8)))  __bf16 v8bf;
typedef __attribute__((ext_vector_type(8)))  float  v8f;
typedef __attribute__((ext_vector_type(4)))  float  v4f;

__device__ __forceinline__ v16bf pack16(v8bf lo, v8bf hi) {
    v16bf r;
#pragma unroll
    for (int i = 0; i < 8; ++i) { r[i] = lo[i]; r[i + 8] = hi[i]; }
    return r;
}

// async DMA: global -> LDS, 128 bits per lane, tracked by ASYNCcnt
__device__ __forceinline__ void async_b128(unsigned int lds_off, const __bf16* g) {
    asm volatile("global_load_async_to_lds_b128 %0, %1, off"
                 :: "v"(lds_off), "v"(g) : "memory");
}
__device__ __forceinline__ void wait_async0() {
    asm volatile("s_wait_asynccnt 0x0" ::: "memory");
}
__device__ __forceinline__ unsigned int lds_off32(const void* p) {
    return (unsigned int)(unsigned long long)p;   // low 32 bits of shared-aperture address
}

// ---------------- f32 -> bf16 bulk convert (hidden_states) ----------------
__global__ __launch_bounds__(256) void k_cvt_bf16(const float* __restrict__ in,
                                                  __bf16* __restrict__ out) {
    int i = (blockIdx.x * 256 + threadIdx.x) * 4;
    v4f v = *(const v4f*)(in + i);
#pragma unroll
    for (int j = 0; j < 4; ++j) out[i + j] = (__bf16)v[j];
}

// ---------------- weight transpose+convert: Wt[n][k] = (bf16)W[k][n] ------
__global__ __launch_bounds__(256) void k_wtrans(const float* __restrict__ W,
                                                __bf16* __restrict__ Wt) {
    int idx = blockIdx.x * 256 + threadIdx.x;   // k*1024 + n
    int n = idx & (H_ - 1);
    int k = idx >> 10;
    Wt[(size_t)n * H_ + k] = (__bf16)W[idx];
}

// ---------------- WMMA GEMM: C[M x 1024] = X(bf16) @ Wt(bf16,[n][k]) + bias
// block tile 64(M) x 128(N), 8 waves -> each wave 16(M) x 64(N), K-step 32.
// Double-buffered LDS staging via global_load_async_to_lds_b128 (ASYNCcnt),
// one barrier per k-step.
__global__ __launch_bounds__(256) void k_gemm(const __bf16* __restrict__ X,
                                              const __bf16* __restrict__ Wt,
                                              const float* __restrict__ bias,
                                              __bf16* __restrict__ obf,
                                              float* __restrict__ of32, int M) {
    __shared__ __bf16 Xs[2][64][32];    //  8 KB
    __shared__ __bf16 Ws[2][128][32];   // 16 KB
    const int tid  = threadIdx.x;
    const int lane = tid & 31;
    const int wave = tid >> 5;
    const int wm   = wave >> 1;      // 0..3
    const int wn   = wave & 1;       // 0..1
    const int lrow = lane & 15;
    const int lhalf = lane >> 4;
    const int m0 = blockIdx.y * 64;
    const int n0 = blockIdx.x * 128;

    // per-thread staging slots: 1 X chunk + 2 W chunks of 8 halves (16B)
    const int sr = tid >> 2;            // 0..63
    const int sc = (tid & 3) * 8;       // 0,8,16,24
    int xr = m0 + sr; if (xr > M - 1) xr = M - 1;   // clamp (stores are guarded)
    const __bf16* xg  = X  + (size_t)xr * H_ + sc;
    const __bf16* wg0 = Wt + (size_t)(n0 + sr) * H_ + sc;
    const __bf16* wg1 = Wt + (size_t)(n0 + sr + 64) * H_ + sc;
    unsigned int xs_off[2], ws0_off[2], ws1_off[2];
#pragma unroll
    for (int b = 0; b < 2; ++b) {
        xs_off[b]  = lds_off32(&Xs[b][sr][sc]);
        ws0_off[b] = lds_off32(&Ws[b][sr][sc]);
        ws1_off[b] = lds_off32(&Ws[b][sr + 64][sc]);
    }

    auto stage = [&](int kidx, int b) {
        async_b128(xs_off[b],  xg  + (size_t)kidx * 32);
        async_b128(ws0_off[b], wg0 + (size_t)kidx * 32);
        async_b128(ws1_off[b], wg1 + (size_t)kidx * 32);
    };

    v8f acc[4] = {};
    const int NK = H_ / 32;             // 32 k-steps
    stage(0, 0);
    for (int j = 0; j < NK; ++j) {
        const int b = j & 1;
        wait_async0();                  // my stage-j DMAs have landed in LDS
        __syncthreads();                // everyone's landed; prev buffer free
        if (j + 1 < NK) stage(j + 1, b ^ 1);   // prefetch overlaps the WMMAs

        // A fragment (16x32 bf16): lane half picks K 0-15 vs 16-31
        const int arow = wm * 16 + lrow;
        const int ab = lhalf * 16;
        v16bf a = pack16(*(const v8bf*)(&Xs[b][arow][ab]),
                         *(const v8bf*)(&Xs[b][arow][ab + 8]));
#pragma unroll
        for (int nt = 0; nt < 4; ++nt) {
            int bn = wn * 64 + nt * 16 + lrow;
            v16bf bf = *(const v16bf*)(&Ws[b][bn][lhalf * 16]);
            acc[nt] = __builtin_amdgcn_wmma_f32_16x16x32_bf16(
                false, a, false, bf, (short)0, acc[nt], false, false);
        }
    }
#pragma unroll
    for (int nt = 0; nt < 4; ++nt) {
        int n = n0 + wn * 64 + nt * 16 + lrow;
        float bb = bias ? bias[n] : 0.0f;
#pragma unroll
        for (int r = 0; r < 8; ++r) {
            int m = m0 + wm * 16 + r + lhalf * 8;
            if (m < M) {
                float v = acc[nt][r] + bb;
                if (of32) of32[(size_t)m * H_ + n] = v;
                else      obf[(size_t)m * H_ + n] = (__bf16)v;
            }
        }
    }
}

// ---------------- RoPE on Q,K (patch tokens t in [5,1029)) ----------------
__global__ __launch_bounds__(256) void k_rope(__bf16* __restrict__ Q,
                                              __bf16* __restrict__ K,
                                              const float* __restrict__ cosp,
                                              const float* __restrict__ sinp) {
    int idx = blockIdx.x * 256 + threadIdx.x;   // B*P*NH*32 threads
    int d = idx & 31;
    int h = (idx >> 5) & 15;
    int p = (idx >> 9) & 1023;
    int b = idx >> 19;
    int t = p + NPREF;
    size_t base = ((size_t)(b * T_ + t)) * H_ + h * 64;
    float c1 = cosp[p * 64 + d],      s1 = sinp[p * 64 + d];
    float c2 = cosp[p * 64 + d + 32], s2 = sinp[p * 64 + d + 32];
    float q1 = (float)Q[base + d], q2 = (float)Q[base + d + 32];
    Q[base + d]      = (__bf16)(q1 * c1 - q2 * s1);
    Q[base + d + 32] = (__bf16)(q2 * c2 + q1 * s2);
    float k1 = (float)K[base + d], k2 = (float)K[base + d + 32];
    K[base + d]      = (__bf16)(k1 * c1 - k2 * s1);
    K[base + d + 32] = (__bf16)(k2 * c2 + k1 * s2);
}

// ---------------- flash attention: per (b,h), 128 Q rows/block ------------
__global__ __launch_bounds__(256) void k_attn(const __bf16* __restrict__ Q,
                                              const __bf16* __restrict__ K,
                                              const __bf16* __restrict__ V,
                                              __bf16* __restrict__ O) {
    __shared__ __bf16 Qs[128][64];   // 16 KB
    __shared__ __bf16 Ks[64][64];    //  8 KB  (row = key, contiguous head-dim)
    __shared__ __bf16 Vts[64][64];   //  8 KB  (row = head-dim, contiguous key)
    __shared__ __bf16 Ps[8][16][64]; // 16 KB  (per-wave P staging)

    const int tid  = threadIdx.x;
    const int lane = tid & 31;
    const int wave = tid >> 5;
    const int lrow = lane & 15;
    const int lhalf = lane >> 4;
    const int b = blockIdx.x >> 4;
    const int h = blockIdx.x & 15;
    const int qt0 = blockIdx.y * 128;
    const size_t rowbase = (size_t)b * T_ * H_ + (size_t)h * 64;

    // stage Q tile (128 rows x 64 dims), zero-fill OOB rows
#pragma unroll
    for (int i = 0; i < 4; ++i) {
        int idx = tid + i * 256;
        int r = idx >> 3, c = (idx & 7) * 8;
        int t = qt0 + r;
        v8bf v = {};
        if (t < T_) v = *(const v8bf*)(Q + rowbase + (size_t)t * H_ + c);
        *(v8bf*)(&Qs[r][c]) = v;
    }
    __syncthreads();

    // per-wave Q A-fragments, loaded once (two 32-wide K windows over HD=64)
    v16bf qa[2];
#pragma unroll
    for (int ks = 0; ks < 2; ++ks) {
        int base0 = ks * 32 + lhalf * 16;
        qa[ks] = pack16(*(const v8bf*)(&Qs[wave * 16 + lrow][base0]),
                        *(const v8bf*)(&Qs[wave * 16 + lrow][base0 + 8]));
    }

    v8f o[4] = {};
    float mrow[8], lsum[8];
#pragma unroll
    for (int r = 0; r < 8; ++r) { mrow[r] = -1e30f; lsum[r] = 0.0f; }

    const float scale = 0.125f;   // 1/sqrt(64)
    for (int j = 0; j < (T_ + 63) / 64; ++j) {
        const int kt0 = j * 64;
        // stage K (row-major) and V (transposed) tiles
#pragma unroll
        for (int i = 0; i < 2; ++i) {
            int idx = tid + i * 256;
            int r = idx >> 3, c = (idx & 7) * 8;
            int t = kt0 + r;
            v8bf kv = {}, vv = {};
            if (t < T_) {
                kv = *(const v8bf*)(K + rowbase + (size_t)t * H_ + c);
                vv = *(const v8bf*)(V + rowbase + (size_t)t * H_ + c);
            }
            *(v8bf*)(&Ks[r][c]) = kv;
#pragma unroll
            for (int q = 0; q < 8; ++q) Vts[c + q][r] = vv[q];
        }
        __syncthreads();

        // scores S = Q @ K^T : 4 tiles of 16x16, K-dim = 64 (2 wmma steps)
        v8f s[4] = {};
#pragma unroll
        for (int nt = 0; nt < 4; ++nt)
#pragma unroll
            for (int ks = 0; ks < 2; ++ks) {
                v16bf bfr = *(const v16bf*)(&Ks[nt * 16 + lrow][ks * 32 + lhalf * 16]);
                s[nt] = __builtin_amdgcn_wmma_f32_16x16x32_bf16(
                    false, qa[ks], false, bfr, (short)0, s[nt], false, false);
            }

        // scale + tail-key mask + online softmax (rows live in VGPR index r
        // within each 16-lane half; reduce across the half with shfl_xor)
        float mnew[8];
#pragma unroll
        for (int r = 0; r < 8; ++r) mnew[r] = mrow[r];
#pragma unroll
        for (int nt = 0; nt < 4; ++nt) {
            bool valid = (kt0 + nt * 16 + lrow) < T_;
#pragma unroll
            for (int r = 0; r < 8; ++r) {
                float v = valid ? s[nt][r] * scale : -1e30f;
                s[nt][r] = v;
                mnew[r] = fmaxf(mnew[r], v);
            }
        }
#pragma unroll
        for (int off = 1; off < 16; off <<= 1)
#pragma unroll
            for (int r = 0; r < 8; ++r)
                mnew[r] = fmaxf(mnew[r], __shfl_xor(mnew[r], off, 32));

        float sf[8], psum[8];
#pragma unroll
        for (int r = 0; r < 8; ++r) { sf[r] = __expf(mrow[r] - mnew[r]); psum[r] = 0.0f; }
#pragma unroll
        for (int nt = 0; nt < 4; ++nt)
#pragma unroll
            for (int r = 0; r < 8; ++r) {
                float p = __expf(s[nt][r] - mnew[r]);
                s[nt][r] = p;
                psum[r] += p;
            }
#pragma unroll
        for (int off = 1; off < 16; off <<= 1)
#pragma unroll
            for (int r = 0; r < 8; ++r) psum[r] += __shfl_xor(psum[r], off, 32);
#pragma unroll
        for (int r = 0; r < 8; ++r) {
            lsum[r] = lsum[r] * sf[r] + psum[r];
            mrow[r] = mnew[r];
        }
        // rescale O accumulators (same row<->VGPR mapping)
#pragma unroll
        for (int dt = 0; dt < 4; ++dt)
#pragma unroll
            for (int r = 0; r < 8; ++r) o[dt][r] *= sf[r];

        // route P from D-layout to A-layout via per-wave LDS (bf16)
#pragma unroll
        for (int nt = 0; nt < 4; ++nt)
#pragma unroll
            for (int r = 0; r < 8; ++r)
                Ps[wave][r + lhalf * 8][nt * 16 + lrow] = (__bf16)s[nt][r];

        // O += P @ V  (K-dim = 64 keys, 2 wmma steps; V^T staged in Vts)
#pragma unroll
        for (int ks = 0; ks < 2; ++ks) {
            int base0 = ks * 32 + lhalf * 16;
            v16bf pa = pack16(*(const v8bf*)(&Ps[wave][lrow][base0]),
                              *(const v8bf*)(&Ps[wave][lrow][base0 + 8]));
#pragma unroll
            for (int dt = 0; dt < 4; ++dt) {
                v16bf vb = *(const v16bf*)(&Vts[dt * 16 + lrow][ks * 32 + lhalf * 16]);
                o[dt] = __builtin_amdgcn_wmma_f32_16x16x32_bf16(
                    false, pa, false, vb, (short)0, o[dt], false, false);
            }
        }
        __syncthreads();
    }

    // normalize and store attention output as bf16 [b*T+t][h*64+d]
#pragma unroll
    for (int dt = 0; dt < 4; ++dt)
#pragma unroll
        for (int r = 0; r < 8; ++r) {
            int t = qt0 + wave * 16 + r + lhalf * 8;
            if (t < T_) {
                float v = o[dt][r] / lsum[r];
                O[rowbase + (size_t)t * H_ + dt * 16 + lrow] = (__bf16)v;
            }
        }
}

// --------------------------------------------------------------------------
extern "C" void kernel_launch(void* const* d_in, const int* in_sizes, int n_in,
                              void* d_out, int out_size, void* d_ws, size_t ws_size,
                              hipStream_t stream) {
    (void)in_sizes; (void)n_in; (void)out_size; (void)ws_size;
    const float* hs   = (const float*)d_in[0];
    const float* cosp = (const float*)d_in[1];
    const float* sinp = (const float*)d_in[2];
    const float* wq   = (const float*)d_in[3];
    const float* bq   = (const float*)d_in[4];
    const float* wk   = (const float*)d_in[5];
    const float* wv   = (const float*)d_in[6];
    const float* bv   = (const float*)d_in[7];
    const float* wo   = (const float*)d_in[8];
    const float* bo   = (const float*)d_in[9];
    float* out = (float*)d_out;

    char* ws = (char*)d_ws;
    size_t off = 0;
    auto alloc = [&](size_t bytes) -> void* {
        void* p = ws + off;
        off += (bytes + 255) & ~(size_t)255;
        return p;
    };
    const size_t actBytes = (size_t)MT_ * H_ * sizeof(__bf16);
    const size_t wBytes   = (size_t)H_ * H_ * sizeof(__bf16);
    __bf16* Xb  = (__bf16*)alloc(actBytes);
    __bf16* Wqt = (__bf16*)alloc(wBytes);
    __bf16* Wkt = (__bf16*)alloc(wBytes);
    __bf16* Wvt = (__bf16*)alloc(wBytes);
    __bf16* Wot = (__bf16*)alloc(wBytes);
    __bf16* Qb  = (__bf16*)alloc(actBytes);
    __bf16* Kb  = (__bf16*)alloc(actBytes);
    __bf16* Vb  = (__bf16*)alloc(actBytes);
    __bf16* Ab  = (__bf16*)alloc(actBytes);

    k_cvt_bf16<<<(MT_ * H_) / 1024, 256, 0, stream>>>(hs, Xb);
    k_wtrans<<<(H_ * H_) / 256, 256, 0, stream>>>(wq, Wqt);
    k_wtrans<<<(H_ * H_) / 256, 256, 0, stream>>>(wk, Wkt);
    k_wtrans<<<(H_ * H_) / 256, 256, 0, stream>>>(wv, Wvt);
    k_wtrans<<<(H_ * H_) / 256, 256, 0, stream>>>(wo, Wot);

    dim3 ggrid(H_ / 128, (MT_ + 63) / 64);
    k_gemm<<<ggrid, 256, 0, stream>>>(Xb, Wqt, bq, Qb, nullptr, MT_);
    k_gemm<<<ggrid, 256, 0, stream>>>(Xb, Wkt, nullptr, Kb, nullptr, MT_);
    k_gemm<<<ggrid, 256, 0, stream>>>(Xb, Wvt, bv, Vb, nullptr, MT_);

    k_rope<<<(B_ * P_ * NH_ * 32) / 256, 256, 0, stream>>>(Qb, Kb, cosp, sinp);

    k_attn<<<dim3(B_ * NH_, (T_ + 127) / 128), 256, 0, stream>>>(Qb, Kb, Vb, Ab);

    k_gemm<<<ggrid, 256, 0, stream>>>(Ab, Wot, bo, nullptr, out, MT_);
}
